// MultiHeadAttention_3693671874986
// MI455X (gfx1250) — compile-verified, matrix operand order checked by reading, not run on hardware
//
#include <hip/hip_runtime.h>
#include <hip/hip_bf16.h>
#include <math.h>

// MI455X / gfx1250: wave32, v_wmma_f32_16x16x32_bf16 (f32 accumulate).

typedef __attribute__((ext_vector_type(16))) __bf16 v16bf;
typedef __attribute__((ext_vector_type(8)))  __bf16 v8bf;
typedef __attribute__((ext_vector_type(8)))  float  v8f;

#define WAVE 32

static constexpr int Bx = 2, Tt = 2048, Cc = 1024, Hh = 16, DK = 64, NC = 3072;

__device__ __forceinline__ v8f wmma_bf16(v16bf a, v16bf b, v8f c) {
    return __builtin_amdgcn_wmma_f32_16x16x32_bf16(
        false, a, false, b, (short)0, c, false, false);
}

// A 16-bit WMMA fragment (A or our column-major B) is, per lane, two
// contiguous 16-byte runs: k in [half*8, half*8+8) and [16+half*8, 16+half*8+8).
// With bf16 operands stored k-contiguously this is two 128-bit loads.
__device__ __forceinline__ v16bf load_frag(const __bf16* base, int half) {
    v8bf c0 = *(const v8bf*)(base + half * 8);
    v8bf c1 = *(const v8bf*)(base + 16 + half * 8);
    return __builtin_shufflevector(c0, c1, 0, 1, 2, 3, 4, 5, 6, 7,
                                           8, 9, 10, 11, 12, 13, 14, 15);
}

__device__ __forceinline__ v8bf cvt8(v8f a) {
    v8bf p;
#pragma unroll
    for (int r = 0; r < 8; ++r) p[r] = (__bf16)a[r];
    return p;
}

// ---------------------------------------------------------------------------
// Preprocess: x f32 -> bf16 (coalesced)
// ---------------------------------------------------------------------------
__global__ __launch_bounds__(256) void cvt_x_kernel(
    const float* __restrict__ x, __bf16* __restrict__ xb)
{
    size_t i = (size_t)blockIdx.x * blockDim.x + threadIdx.x;
    float2 f = ((const float2*)x)[i];
    xb[2 * i]     = (__bf16)f.x;
    xb[2 * i + 1] = (__bf16)f.y;
}

// Preprocess: weight (R x N, f32, row-major) -> n-major bf16 (N x R),
// LDS-tiled 16x16 transpose, coalesced on both sides.
__global__ __launch_bounds__(256) void transpose_cvt_kernel(
    const float* __restrict__ src, __bf16* __restrict__ dst, int R, int N)
{
    __shared__ float tile[16][17];
    const int tx = threadIdx.x & 15, ty = threadIdx.x >> 4;
    const int c0 = blockIdx.x * 16, r0 = blockIdx.y * 16;
    tile[ty][tx] = src[(size_t)(r0 + ty) * N + c0 + tx];
    __syncthreads();
    dst[(size_t)(c0 + ty) * R + r0 + tx] = (__bf16)tile[tx][ty];
}

// ---------------------------------------------------------------------------
// Kernel 1: qkv = x @ w_qkv. One wave computes a 16x64 strip (4 WMMA tiles,
// A fragment reused 4x). Scatters to Q*scale, K (B,H,T,dk) and V^T (B,H,dk,T).
// ---------------------------------------------------------------------------
__global__ __launch_bounds__(128) void qkv_gemm_kernel(
    const __bf16* __restrict__ xb, const __bf16* __restrict__ wqkvT,
    __bf16* __restrict__ qb, __bf16* __restrict__ kb, __bf16* __restrict__ vtb)
{
    const int wave = (blockIdx.x * blockDim.x + threadIdx.x) >> 5;
    const int lane = threadIdx.x & 31;
    const int ntiles = NC / 64;               // 48 strips of 64 cols
    const int mt = wave / ntiles;
    const int nt = wave % ntiles;
    const int m0 = mt * 16, n0 = nt * 64;
    const int half = lane >> 4, l16 = lane & 15;

    const __bf16* arow = xb + (size_t)(m0 + l16) * Cc;
    const __bf16* brow0 = wqkvT + (size_t)(n0 +  0 + l16) * Cc;
    const __bf16* brow1 = wqkvT + (size_t)(n0 + 16 + l16) * Cc;
    const __bf16* brow2 = wqkvT + (size_t)(n0 + 32 + l16) * Cc;
    const __bf16* brow3 = wqkvT + (size_t)(n0 + 48 + l16) * Cc;

    v8f acc0 = {}, acc1 = {}, acc2 = {}, acc3 = {};
    for (int k0 = 0; k0 < Cc; k0 += 32) {
        __builtin_prefetch(arow + k0 + 128, 0, 1);   // global_prefetch_b8
        const v16bf a = load_frag(arow + k0, half);
        acc0 = wmma_bf16(a, load_frag(brow0 + k0, half), acc0);
        acc1 = wmma_bf16(a, load_frag(brow1 + k0, half), acc1);
        acc2 = wmma_bf16(a, load_frag(brow2 + k0, half), acc2);
        acc3 = wmma_bf16(a, load_frag(brow3 + k0, half), acc3);
    }

    // Wave-uniform routing: a 64-wide strip never crosses a q/k/v boundary
    // and maps to a single head.
    const int w  = n0 >> 10;                 // 0=q, 1=k, 2=v
    const int h  = (n0 & 1023) >> 6;
    const int bb = m0 / Tt;
    const int trow = (m0 % Tt) + half * 8;   // 8 consecutive t per lane

    if (w == 2) {
        // V^T: t is contiguous across the 8 accumulator rows -> b128 stores.
        __bf16* vbase = vtb + (size_t)(bb * Hh + h) * DK * Tt;
        *(v8bf*)(vbase + (size_t)( 0 + l16) * Tt + trow) = cvt8(acc0);
        *(v8bf*)(vbase + (size_t)(16 + l16) * Tt + trow) = cvt8(acc1);
        *(v8bf*)(vbase + (size_t)(32 + l16) * Tt + trow) = cvt8(acc2);
        *(v8bf*)(vbase + (size_t)(48 + l16) * Tt + trow) = cvt8(acc3);
    } else {
        // Q is pre-scaled by 1/sqrt(dk)=0.125 (power of two: exact in bf16).
        __bf16* qk = (w == 0 ? qb : kb) + (size_t)(bb * Hh + h) * Tt * DK;
        const float sc = (w == 0) ? 0.125f : 1.0f;
#pragma unroll
        for (int r = 0; r < 8; ++r) {
            const size_t t = trow + r;
            qk[t * DK +  0 + l16] = (__bf16)(acc0[r] * sc);
            qk[t * DK + 16 + l16] = (__bf16)(acc1[r] * sc);
            qk[t * DK + 32 + l16] = (__bf16)(acc2[r] * sc);
            qk[t * DK + 48 + l16] = (__bf16)(acc3[r] * sc);
        }
    }
}

// ---------------------------------------------------------------------------
// Kernel 2: causal flash attention. One wave per (b,h, 16-row q-tile).
// Unmasked blocks skip the causal compare; only diagonal blocks pay it.
// ---------------------------------------------------------------------------
__global__ __launch_bounds__(128) void flash_attn_kernel(
    const __bf16* __restrict__ qb, const __bf16* __restrict__ kb,
    const __bf16* __restrict__ vtb, __bf16* __restrict__ ob)
{
    __shared__ __align__(16) __bf16 lds_p[4][16][32];

    const int lane = threadIdx.x & 31;
    const int wv   = threadIdx.x >> 5;
    const int gw   = blockIdx.x * 4 + wv;
    const int qtiles = Tt / 16;
    const int bh = gw / qtiles;
    const int qt = gw % qtiles;
    const int bb = bh / Hh, h = bh % Hh;
    const int q0 = qt * 16;
    const int half = lane >> 4, l16 = lane & 15;

    const __bf16* Q  = qb  + (size_t)bh * Tt * DK;
    const __bf16* K  = kb  + (size_t)bh * Tt * DK;
    const __bf16* Vt = vtb + (size_t)bh * DK * Tt;

    const __bf16* qrow = Q + (size_t)(q0 + l16) * DK;
    const v16bf aq0 = load_frag(qrow, half);
    const v16bf aq1 = load_frag(qrow + 32, half);

    v8f o0 = {}, o1 = {}, o2 = {}, o3 = {};
    float m_r[8], l_r[8];
#pragma unroll
    for (int r = 0; r < 8; ++r) { m_r[r] = -1e30f; l_r[r] = 0.f; }
    const float L2E = 1.44269504f;

    auto process_block = [&](int k0, bool domask) {
        const __bf16* kr0 = K + (size_t)(k0 + l16) * DK;
        const __bf16* kr1 = K + (size_t)(k0 + 16 + l16) * DK;
        const v16bf bk00 = load_frag(kr0, half);
        const v16bf bk01 = load_frag(kr0 + 32, half);
        const v16bf bk10 = load_frag(kr1, half);
        const v16bf bk11 = load_frag(kr1 + 32, half);

        v8f s0 = {}, s1 = {};
        s0 = wmma_bf16(aq0, bk00, s0);
        s0 = wmma_bf16(aq1, bk01, s0);
        s1 = wmma_bf16(aq0, bk10, s1);
        s1 = wmma_bf16(aq1, bk11, s1);
        // Q was pre-scaled by 1/sqrt(dk); scores are final logits.

        if (domask) {
#pragma unroll
            for (int r = 0; r < 8; ++r) {
                const int row = q0 + r + half * 8;
                if (k0 + l16 > row)      s0[r] = -1e30f;
                if (k0 + 16 + l16 > row) s1[r] = -1e30f;
            }
        }

        // Online softmax: each row lives across the 16 lanes of a half-group.
#pragma unroll
        for (int r = 0; r < 8; ++r) {
            float bm = fmaxf(s0[r], s1[r]);
            bm = fmaxf(bm, __shfl_xor(bm, 1, WAVE));
            bm = fmaxf(bm, __shfl_xor(bm, 2, WAVE));
            bm = fmaxf(bm, __shfl_xor(bm, 4, WAVE));
            bm = fmaxf(bm, __shfl_xor(bm, 8, WAVE));
            const float mn    = fmaxf(m_r[r], bm);
            const float alpha = exp2f((m_r[r] - mn) * L2E);
            const float p0 = exp2f((s0[r] - mn) * L2E);
            const float p1 = exp2f((s1[r] - mn) * L2E);
            float rs = p0 + p1;
            rs += __shfl_xor(rs, 1, WAVE);
            rs += __shfl_xor(rs, 2, WAVE);
            rs += __shfl_xor(rs, 4, WAVE);
            rs += __shfl_xor(rs, 8, WAVE);
            l_r[r] = l_r[r] * alpha + rs;
            m_r[r] = mn;
            o0[r] *= alpha; o1[r] *= alpha; o2[r] *= alpha; o3[r] *= alpha;
            lds_p[wv][r + half * 8][l16]      = (__bf16)p0;
            lds_p[wv][r + half * 8][16 + l16] = (__bf16)p1;
        }

        // Reload P as A fragment (ds_load_b128 x2), V^T rows are k-contiguous.
        const v16bf ap = load_frag(&lds_p[wv][l16][0], half);
        const v16bf bv0 = load_frag(Vt + (size_t)( 0 + l16) * Tt + k0, half);
        const v16bf bv1 = load_frag(Vt + (size_t)(16 + l16) * Tt + k0, half);
        const v16bf bv2 = load_frag(Vt + (size_t)(32 + l16) * Tt + k0, half);
        const v16bf bv3 = load_frag(Vt + (size_t)(48 + l16) * Tt + k0, half);
        o0 = wmma_bf16(ap, bv0, o0);
        o1 = wmma_bf16(ap, bv1, o1);
        o2 = wmma_bf16(ap, bv2, o2);
        o3 = wmma_bf16(ap, bv3, o3);
    };

    const int kend  = q0 + 16;                  // causal bound (exclusive)
    const int kfull = ((q0 + 1) >> 5) << 5;     // blocks fully below diagonal
    int k0 = 0;
    for (; k0 < kfull; k0 += 32) process_block(k0, false);
    for (; k0 < kend;  k0 += 32) process_block(k0, true);

    // Normalize and store attention output (B, T, C) bf16, heads packed.
#pragma unroll
    for (int r = 0; r < 8; ++r) {
        const int row = q0 + r + half * 8;
        const float inv = 1.0f / l_r[r];
        const size_t base = ((size_t)bb * Tt + row) * Cc + h * DK;
        ob[base +  0 + l16] = (__bf16)(o0[r] * inv);
        ob[base + 16 + l16] = (__bf16)(o1[r] * inv);
        ob[base + 32 + l16] = (__bf16)(o2[r] * inv);
        ob[base + 48 + l16] = (__bf16)(o3[r] * inv);
    }
}

// ---------------------------------------------------------------------------
// Kernel 3: out = attn_out @ w_out + b_out (f32 result). 16x64 per wave.
// ---------------------------------------------------------------------------
__global__ __launch_bounds__(128) void out_gemm_kernel(
    const __bf16* __restrict__ ao, const __bf16* __restrict__ woutT,
    const float* __restrict__ bout, float* __restrict__ out)
{
    const int wave = (blockIdx.x * blockDim.x + threadIdx.x) >> 5;
    const int lane = threadIdx.x & 31;
    const int ntiles = Cc / 64;               // 16 strips
    const int mt = wave / ntiles;
    const int nt = wave % ntiles;
    const int m0 = mt * 16, n0 = nt * 64;
    const int half = lane >> 4, l16 = lane & 15;

    const __bf16* arow = ao + (size_t)(m0 + l16) * Cc;
    const __bf16* brow0 = woutT + (size_t)(n0 +  0 + l16) * Cc;
    const __bf16* brow1 = woutT + (size_t)(n0 + 16 + l16) * Cc;
    const __bf16* brow2 = woutT + (size_t)(n0 + 32 + l16) * Cc;
    const __bf16* brow3 = woutT + (size_t)(n0 + 48 + l16) * Cc;

    v8f acc0 = {}, acc1 = {}, acc2 = {}, acc3 = {};
    for (int k0 = 0; k0 < Cc; k0 += 32) {
        __builtin_prefetch(arow + k0 + 128, 0, 1);
        const v16bf a = load_frag(arow + k0, half);
        acc0 = wmma_bf16(a, load_frag(brow0 + k0, half), acc0);
        acc1 = wmma_bf16(a, load_frag(brow1 + k0, half), acc1);
        acc2 = wmma_bf16(a, load_frag(brow2 + k0, half), acc2);
        acc3 = wmma_bf16(a, load_frag(brow3 + k0, half), acc3);
    }

    const float bi0 = bout[n0 +  0 + l16];
    const float bi1 = bout[n0 + 16 + l16];
    const float bi2 = bout[n0 + 32 + l16];
    const float bi3 = bout[n0 + 48 + l16];
#pragma unroll
    for (int r = 0; r < 8; ++r) {
        const size_t m = m0 + r + half * 8;
        out[m * Cc + n0 +  0 + l16] = acc0[r] + bi0;
        out[m * Cc + n0 + 16 + l16] = acc1[r] + bi1;
        out[m * Cc + n0 + 32 + l16] = acc2[r] + bi2;
        out[m * Cc + n0 + 48 + l16] = acc3[r] + bi3;
    }
}

// ---------------------------------------------------------------------------
extern "C" void kernel_launch(void* const* d_in, const int* in_sizes, int n_in,
                              void* d_out, int out_size, void* d_ws, size_t ws_size,
                              hipStream_t stream) {
    (void)in_sizes; (void)n_in; (void)out_size; (void)ws_size;
    const float* x    = (const float*)d_in[0];
    // d_in[1] = causal mask (tril) — realized via index compare in-kernel.
    const float* wqkv = (const float*)d_in[2];
    const float* wout = (const float*)d_in[3];
    const float* bout = (const float*)d_in[4];
    float* out = (float*)d_out;

    // Workspace (bf16): xb, wqkvT, woutT, Q, K, V^T, attn_out  (~48 MB).
    const size_t nx  = (size_t)Bx * Tt * Cc;      // 4,194,304
    const size_t nq  = (size_t)Bx * Hh * Tt * DK; // 4,194,304
    __bf16* xb     = (__bf16*)d_ws;
    __bf16* wqkvT  = xb + nx;
    __bf16* woutT  = wqkvT + (size_t)Cc * NC;
    __bf16* qb     = woutT + (size_t)Cc * Cc;
    __bf16* kb     = qb + nq;
    __bf16* vtb    = kb + nq;
    __bf16* ob     = vtb + nq;

    // Preprocess: convert + transpose (all cheap, bandwidth-friendly).
    cvt_x_kernel<<<(int)(nx / 2 / 256), 256, 0, stream>>>(x, xb);
    transpose_cvt_kernel<<<dim3(NC / 16, Cc / 16), 256, 0, stream>>>(wqkv, wqkvT, Cc, NC);
    transpose_cvt_kernel<<<dim3(Cc / 16, Cc / 16), 256, 0, stream>>>(wout, woutT, Cc, Cc);

    // 1) QKV projection: (4096/16) * (3072/64) = 12288 waves, 4/block.
    qkv_gemm_kernel<<<(Bx * Tt / 16) * (NC / 64) / 4, 128, 0, stream>>>(
        xb, wqkvT, qb, kb, vtb);
    // 2) Flash attention: B*H*(T/16) = 4096 waves, 4/block.
    flash_attn_kernel<<<Bx * Hh * (Tt / 16) / 4, 128, 0, stream>>>(
        qb, kb, vtb, ob);
    // 3) Output projection: (4096/16) * (1024/64) = 4096 waves, 4/block.
    out_gemm_kernel<<<(Bx * Tt / 16) * (Cc / 64) / 4, 128, 0, stream>>>(
        ob, woutT, bout, out);
}
